// StackedBidirectionalLstm_86603720556802
// MI455X (gfx1250) — matrix operand order repaired and checked by reading, not compiled, not run on hardware
//
#include <hip/hip_runtime.h>
#include <hip/hip_bf16.h>

// ---------------------------------------------------------------------------
// Stacked bidirectional augmented (highway) LSTM for MI455X / gfx1250.
//   Phase 1: px = x @ Wx as dense bf16 WMMA GEMM (full-device parallel).
//   Phase 2: time recurrence ps = h @ Wh with wave32 WMMA tiles across 20
//            cooperating workgroups + device-scope spin barrier (2/step).
//            h is staged per-block in LDS (conflict-free padded stride) and
//            next-step px operands are prefetched (global_prefetch_b8).
// Workspace requirement: ~282 MB (bf16 weights, two bf16 px buffers, state).
// ---------------------------------------------------------------------------

#define DEVINL __device__ __forceinline__

typedef __attribute__((ext_vector_type(16))) __bf16 v16bf;
typedef __attribute__((ext_vector_type(8)))  float  v8f;

union Frag16 { v16bf v; uint4 q[2]; };   // 32 bytes: one wave32 16-bit WMMA operand (8 VGPRs/lane)

static constexpr int Bsz = 32, Tsz = 512, INsz = 512, Hsz = 512;
static constexpr int G5 = 5 * Hsz;            // 2560 (i,f,g,o,r)
static constexpr int G6 = 6 * Hsz;            // 3072 (+highway)
static constexpr int NBLK_DIR = 10;           // workgroups per direction in recurrence
static constexpr int LSTM_BLOCKS = 2 * NBLK_DIR;
static constexpr int LSTM_THREADS = 512;      // 16 wave32 waves
static constexpr int LDA_LDS = Hsz + 8;       // 520 bf16 = 260 dwords; 260%64=4 -> lanes m hit
                                              // banks 4m..4m+3: all 64 banks once, no conflicts
static constexpr long long OUT0 = (long long)Bsz * Tsz * 2 * Hsz;  // 16,777,216 floats

// ---- bf16 helpers (RNE) ----------------------------------------------------
DEVINL __bf16 f2bf(float f) {
  union { float f; unsigned u; } v; v.f = f;
  unsigned r = (v.u + 0x7FFFu + ((v.u >> 16) & 1u)) >> 16;
  union { unsigned short s; __bf16 b; } o; o.s = (unsigned short)r;
  return o.b;
}
DEVINL float bf2f(__bf16 b) {
  union { unsigned short s; __bf16 b; } i; i.b = b;
  union { unsigned u; float f; } o; o.u = ((unsigned)i.s) << 16;
  return o.f;
}
DEVINL float sigmoidf(float x) { return 1.0f / (1.0f + __expf(-x)); }

// ---- small utility kernels -------------------------------------------------
__global__ void zero_u32_kernel(unsigned* __restrict__ p, int n) {
  int i = blockIdx.x * blockDim.x + threadIdx.x;
  if (i < n) p[i] = 0u;
}

// x (B,T,IN) f32  ->  xbf (T,B,IN) bf16   (time-major rows for the px GEMM)
__global__ void pack_x_kernel(const float* __restrict__ x, __bf16* __restrict__ xbf) {
  long long e = (long long)blockIdx.x * blockDim.x + threadIdx.x;
  long long total = (long long)Bsz * Tsz * INsz;
  if (e >= total) return;
  int i = (int)(e % INsz);
  long long bt = e / INsz;
  int t = (int)(bt % Tsz);
  int b = (int)(bt / Tsz);
  xbf[((long long)t * Bsz + b) * INsz + i] = f2bf(x[e]);
}

// src (K,N) f32  ->  dst (N,K) bf16   (N-major so B fragments are contiguous)
__global__ void transpose_bf16_kernel(const float* __restrict__ src,
                                      __bf16* __restrict__ dst, int K, int N) {
  long long e = (long long)blockIdx.x * blockDim.x + threadIdx.x;
  if (e >= (long long)K * N) return;
  int n = (int)(e % N);
  long long k = e / N;
  dst[(long long)n * K + k] = f2bf(src[e]);
}

// ---- px = A(16384 x K) @ W(K x 3072), bf16 in, f32 acc, bf16 out -----------
// Wave32 fragment layouts per CDNA5 ISA 7.12.2:
//   A 16x32 bf16: lane half selects K-octet; element j -> K = (j/8)*16 + 8*half + (j%8)
//   B 32x16 bf16 (via N-major W^T rows): element j -> K = (j/8)*16 + 8*half + (j%8)
//   C/D 16x16 f32: reg r -> M = r + 8*half, N = lane&15
__global__ void __launch_bounds__(256)
gemm_px_kernel(const __bf16* __restrict__ Ain, const __bf16* __restrict__ Wt,
               __bf16* __restrict__ px, int K) {
  const int lane = threadIdx.x & 31;
  const int wave = threadIdx.x >> 5;             // 0..7
  const int m    = lane & 15;
  const int half = lane >> 4;
  const int mt   = blockIdx.x;                   // 0..1023 (M tiles of 16 rows)
  const int ntBase = blockIdx.y * 32 + wave * 4; // 4 N-tiles per wave, 192 total

  const __bf16* Arow = Ain + (long long)(mt * 16 + m) * K;
  v8f acc[4];
#pragma unroll
  for (int u = 0; u < 4; ++u) acc[u] = (v8f){0.f,0.f,0.f,0.f,0.f,0.f,0.f,0.f};

  for (int k0 = 0; k0 < K; k0 += 32) {
    Frag16 a;
    a.q[0] = *(const uint4*)(Arow + k0 + half * 8);
    a.q[1] = *(const uint4*)(Arow + k0 + 16 + half * 8);
#pragma unroll
    for (int u = 0; u < 4; ++u) {
      const __bf16* Brow = Wt + (long long)((ntBase + u) * 16 + m) * K + k0 + half * 8;
      Frag16 b;
      b.q[0] = *(const uint4*)(Brow);
      b.q[1] = *(const uint4*)(Brow + 16);
      acc[u] = __builtin_amdgcn_wmma_f32_16x16x32_bf16(
          false, a.v, false, b.v, (short)0, acc[u], false, false);
    }
  }
#pragma unroll
  for (int u = 0; u < 4; ++u) {
#pragma unroll
    for (int r = 0; r < 8; ++r) {
      int row = mt * 16 + r + 8 * half;
      int col = (ntBase + u) * 16 + m;
      px[(long long)row * G6 + col] = f2bf(acc[u][r]);
    }
  }
}

// ---- device-scope arrive/spin barrier (all LSTM_BLOCKS blocks resident) ----
DEVINL void grid_barrier(unsigned* bar) {
  __threadfence();
  __syncthreads();
  if (threadIdx.x == 0) {
    unsigned ticket = __hip_atomic_fetch_add(bar, 1u, __ATOMIC_ACQ_REL,
                                             __HIP_MEMORY_SCOPE_AGENT);
    unsigned target = (ticket / (unsigned)LSTM_BLOCKS + 1u) * (unsigned)LSTM_BLOCKS;
    while (__hip_atomic_load(bar, __ATOMIC_ACQUIRE, __HIP_MEMORY_SCOPE_AGENT) < target) {
      __builtin_amdgcn_s_sleep(2);
    }
  }
  __syncthreads();
  __threadfence();
}

// ---- one bidirectional LSTM layer (both directions concurrently) -----------
// Each direction: 10 blocks x 16 waves = 160 waves; wave w owns N-tile w and
// both M-tiles (rows 0-15, 16-31). Per step:
//   stage h (32x512 bf16) global -> LDS once per block (ds_store_b128),
//   16 K-iters x 2 WMMA per wave with A-fragments from LDS (ds_load_b128),
//   ps written f32; barrier; gate pass over the 32x512 hidden elements (cell
//   state in registers, next-step px prefetched); barrier.
__global__ void __launch_bounds__(LSTM_THREADS)
lstm_layer_kernel(const __bf16* __restrict__ pxF, const __bf16* __restrict__ pxB,
                  const __bf16* __restrict__ WhF, const __bf16* __restrict__ WhB,
                  const float* __restrict__ bF, const float* __restrict__ bB,
                  __bf16* __restrict__ hbf,   // 2 * B * H bf16 (zeroed before launch)
                  float* __restrict__ psbuf,  // 2 * B * 5H f32 scratch
                  unsigned* __restrict__ bar, // zeroed before launch
                  __bf16* __restrict__ ybf,   // layer-0 out (T,B,2H) bf16, or null
                  float* __restrict__ yout,   // layer-1 out (B,T,2H) f32, or null
                  float* __restrict__ finalh, float* __restrict__ finalc) {
  __shared__ __bf16 hls[Bsz * LDA_LDS];       // 33,280 B of the 320 KB WGP LDS

  const int dir    = blockIdx.x / NBLK_DIR;
  const int blk    = blockIdx.x % NBLK_DIR;
  const int lane   = threadIdx.x & 31;
  const int waveId = threadIdx.x >> 5;        // 0..15
  const int m      = lane & 15;
  const int half   = lane >> 4;
  const int nt     = blk * 16 + waveId;       // 0..159 (N tiles of 2560 cols)

  const __bf16* px   = dir ? pxB : pxF;
  const __bf16* Wh   = dir ? WhB : WhF;
  const float*  bias = dir ? bB  : bF;
  __bf16* hd  = hbf   + (long long)dir * Bsz * Hsz;
  float*  psd = psbuf + (long long)dir * Bsz * G5;

  const int dth     = blk * LSTM_THREADS + threadIdx.x;  // 0..5119 within dir
  const int dstride = NBLK_DIR * LSTM_THREADS;           // 5120

  const __bf16* Brow = Wh + (long long)(nt * 16 + m) * Hsz;
  const __bf16* Als0 = &hls[m * LDA_LDS];                // M tile 0 (rows 0-15)
  const __bf16* Als1 = &hls[(16 + m) * LDA_LDS];         // M tile 1 (rows 16-31)

  float creg[4] = {0.f, 0.f, 0.f, 0.f};                  // cell state, fixed map

  for (int step = 0; step < Tsz; ++step) {
    const int t = dir ? (Tsz - 1 - step) : step;

    // -------- stage h_{t-1} into LDS (32 KB, one pass per block) --------
    {
      const uint4* src = (const uint4*)hd;               // 2048 x 16B chunks
#pragma unroll
      for (int c = threadIdx.x; c < (Bsz * Hsz) / 8; c += LSTM_THREADS) {
        int row = c >> 6;                                // /64 chunks per row
        int col = c & 63;
        *(uint4*)(&hls[row * LDA_LDS + col * 8]) = src[c];
      }
    }
    __syncthreads();

    // -------- recurrent GEMM: ps = h_{t-1} @ Wh (A from LDS) --------
    v8f c0 = (v8f){0.f,0.f,0.f,0.f,0.f,0.f,0.f,0.f};
    v8f c1 = (v8f){0.f,0.f,0.f,0.f,0.f,0.f,0.f,0.f};
    for (int k0 = 0; k0 < Hsz; k0 += 32) {
      Frag16 b;
      b.q[0] = *(const uint4*)(Brow + k0 + half * 8);
      b.q[1] = *(const uint4*)(Brow + k0 + 16 + half * 8);
      Frag16 a0, a1;
      a0.q[0] = *(const uint4*)(Als0 + k0 + half * 8);
      a0.q[1] = *(const uint4*)(Als0 + k0 + 16 + half * 8);
      a1.q[0] = *(const uint4*)(Als1 + k0 + half * 8);
      a1.q[1] = *(const uint4*)(Als1 + k0 + 16 + half * 8);
      c0 = __builtin_amdgcn_wmma_f32_16x16x32_bf16(false, a0.v, false, b.v,
                                                   (short)0, c0, false, false);
      c1 = __builtin_amdgcn_wmma_f32_16x16x32_bf16(false, a1.v, false, b.v,
                                                   (short)0, c1, false, false);
    }
    {
      const int n = nt * 16 + m;
#pragma unroll
      for (int r = 0; r < 8; ++r) {
        psd[(long long)(r + 8 * half) * G5 + n]      = c0[r];
        psd[(long long)(16 + r + 8 * half) * G5 + n] = c1[r];
      }
    }
    grid_barrier(bar);

    // -------- gates / state update / outputs --------
    const int tn = dir ? (t - 1) : (t + 1);              // next step's time index
    int slot = 0;
    for (int e = dth; e < Bsz * Hsz; e += dstride, ++slot) {
      const int b = e >> 9;              // /512
      const int j = e & (Hsz - 1);
      const float* ps = psd + (long long)b * G5;
      const long long pxr = ((long long)t * Bsz + b) * G6;
      if (tn >= 0 && tn < Tsz) {         // hide HBM latency of next step's px
        const __bf16* pn = px + ((long long)tn * Bsz + b) * G6 + j;
#pragma unroll
        for (int g = 0; g < 6; ++g) __builtin_prefetch(pn + g * Hsz, 0, 1);
      }
      float pi = bf2f(px[pxr + j])           + ps[j]           + bias[j];
      float pf = bf2f(px[pxr + Hsz + j])     + ps[Hsz + j]     + bias[Hsz + j];
      float pg = bf2f(px[pxr + 2 * Hsz + j]) + ps[2 * Hsz + j] + bias[2 * Hsz + j];
      float po = bf2f(px[pxr + 3 * Hsz + j]) + ps[3 * Hsz + j] + bias[3 * Hsz + j];
      float pr = bf2f(px[pxr + 4 * Hsz + j]) + ps[4 * Hsz + j] + bias[4 * Hsz + j];
      float hw = bf2f(px[pxr + 5 * Hsz + j]);                  // highway input
      float ig = sigmoidf(pi), fg = sigmoidf(pf), og = sigmoidf(po), rg = sigmoidf(pr);
      float gg = tanhf(pg);
      float cn = ig * gg + fg * creg[slot];
      creg[slot] = cn;
      float ht = og * tanhf(cn);
      ht = rg * ht + (1.0f - rg) * hw;
      hd[e] = f2bf(ht);
      if (ybf)  ybf[((long long)t * Bsz + b) * (2 * Hsz) + dir * Hsz + j] = f2bf(ht);
      if (yout) yout[((long long)b * Tsz + t) * (2 * Hsz) + dir * Hsz + j] = ht;
      if (step == Tsz - 1) {
        finalh[(long long)b * (2 * Hsz) + dir * Hsz + j] = ht;
        finalc[(long long)b * (2 * Hsz) + dir * Hsz + j] = cn;
      }
    }
    grid_barrier(bar);
  }
}

// ---------------------------------------------------------------------------
extern "C" void kernel_launch(void* const* d_in, const int* in_sizes, int n_in,
                              void* d_out, int out_size, void* d_ws, size_t ws_size,
                              hipStream_t stream) {
  (void)in_sizes; (void)n_in; (void)out_size; (void)ws_size;
  const float* x    = (const float*)d_in[0];
  const float* Wx0f = (const float*)d_in[1];
  const float* Wh0f = (const float*)d_in[2];
  const float* b0f  = (const float*)d_in[3];
  const float* Wx0b = (const float*)d_in[4];
  const float* Wh0b = (const float*)d_in[5];
  const float* b0b  = (const float*)d_in[6];
  const float* Wx1f = (const float*)d_in[7];
  const float* Wh1f = (const float*)d_in[8];
  const float* b1f  = (const float*)d_in[9];
  const float* Wx1b = (const float*)d_in[10];
  const float* Wh1b = (const float*)d_in[11];
  const float* b1b  = (const float*)d_in[12];

  float* out    = (float*)d_out;
  float* finalh = out + OUT0;                         // (2*B, 2H) = 65536 floats
  float* finalc = finalh + (long long)2 * Bsz * 2 * Hsz;

  char* ws = (char*)d_ws;
  size_t off = 0;
  auto alloc = [&](size_t bytes) -> size_t {
    size_t o = off; off = (off + bytes + 255) & ~(size_t)255; return o;
  };
  size_t o_xbf  = alloc((size_t)Tsz * Bsz * INsz * 2);        // 16 MB
  size_t o_y1   = alloc((size_t)Tsz * Bsz * 2 * Hsz * 2);     // 32 MB
  size_t o_wx0f = alloc((size_t)G6 * INsz * 2);
  size_t o_wx0b = alloc((size_t)G6 * INsz * 2);
  size_t o_wx1f = alloc((size_t)G6 * 2 * Hsz * 2);
  size_t o_wx1b = alloc((size_t)G6 * 2 * Hsz * 2);
  size_t o_wh0f = alloc((size_t)G5 * Hsz * 2);
  size_t o_wh0b = alloc((size_t)G5 * Hsz * 2);
  size_t o_wh1f = alloc((size_t)G5 * Hsz * 2);
  size_t o_wh1b = alloc((size_t)G5 * Hsz * 2);
  size_t o_pxF  = alloc((size_t)Tsz * Bsz * G6 * 2);          // 96 MB
  size_t o_pxB  = alloc((size_t)Tsz * Bsz * G6 * 2);          // 96 MB
  size_t o_bar  = alloc(256);                                 // barrier counter
  size_t o_hbf  = alloc((size_t)2 * Bsz * Hsz * 2);           // h state (bf16, 2 dirs)
  size_t o_ps   = alloc((size_t)2 * Bsz * G5 * 4);            // ps scratch f32

  __bf16* xbf  = (__bf16*)(ws + o_xbf);
  __bf16* y1   = (__bf16*)(ws + o_y1);
  __bf16* wx0f = (__bf16*)(ws + o_wx0f);
  __bf16* wx0b = (__bf16*)(ws + o_wx0b);
  __bf16* wx1f = (__bf16*)(ws + o_wx1f);
  __bf16* wx1b = (__bf16*)(ws + o_wx1b);
  __bf16* wh0f = (__bf16*)(ws + o_wh0f);
  __bf16* wh0b = (__bf16*)(ws + o_wh0b);
  __bf16* wh1f = (__bf16*)(ws + o_wh1f);
  __bf16* wh1b = (__bf16*)(ws + o_wh1b);
  __bf16* pxF  = (__bf16*)(ws + o_pxF);
  __bf16* pxB  = (__bf16*)(ws + o_pxB);
  unsigned* bar = (unsigned*)(ws + o_bar);
  __bf16* hbf  = (__bf16*)(ws + o_hbf);
  float*  psb  = (float*)(ws + o_ps);

  // --- pack inputs / weights to bf16 (transposed N-major for B fragments) ---
  {
    long long n = (long long)Bsz * Tsz * INsz;
    pack_x_kernel<<<(unsigned)((n + 255) / 256), 256, 0, stream>>>(x, xbf);
  }
  auto tr = [&](const float* src, __bf16* dst, int K, int N) {
    long long n = (long long)K * N;
    transpose_bf16_kernel<<<(unsigned)((n + 255) / 256), 256, 0, stream>>>(src, dst, K, N);
  };
  tr(Wx0f, wx0f, INsz, G6);     tr(Wx0b, wx0b, INsz, G6);
  tr(Wx1f, wx1f, 2 * Hsz, G6);  tr(Wx1b, wx1b, 2 * Hsz, G6);
  tr(Wh0f, wh0f, Hsz, G5);      tr(Wh0b, wh0b, Hsz, G5);
  tr(Wh1f, wh1f, Hsz, G5);      tr(Wh1b, wh1b, Hsz, G5);

  const dim3 gpx(1024, 6);  // 1024 M-tiles x (6*8 waves * 4) = 192 N-tiles
  const int nz = (int)((256 + (size_t)2 * Bsz * Hsz * 2) / 4);  // bar + hbf

  // --- layer 0 ---
  gemm_px_kernel<<<gpx, 256, 0, stream>>>(xbf, wx0f, pxF, INsz);
  gemm_px_kernel<<<gpx, 256, 0, stream>>>(xbf, wx0b, pxB, INsz);
  zero_u32_kernel<<<(nz + 255) / 256, 256, 0, stream>>>((unsigned*)(ws + o_bar), nz);
  lstm_layer_kernel<<<LSTM_BLOCKS, LSTM_THREADS, 0, stream>>>(
      pxF, pxB, wh0f, wh0b, b0f, b0b, hbf, psb, bar,
      y1, nullptr, finalh, finalc);

  // --- layer 1 ---
  gemm_px_kernel<<<gpx, 256, 0, stream>>>(y1, wx1f, pxF, 2 * Hsz);
  gemm_px_kernel<<<gpx, 256, 0, stream>>>(y1, wx1b, pxB, 2 * Hsz);
  zero_u32_kernel<<<(nz + 255) / 256, 256, 0, stream>>>((unsigned*)(ws + o_bar), nz);
  lstm_layer_kernel<<<LSTM_BLOCKS, LSTM_THREADS, 0, stream>>>(
      pxF, pxB, wh1f, wh1b, b1f, b1b, hbf, psb, bar,
      nullptr, out, finalh + (long long)Bsz * 2 * Hsz, finalc + (long long)Bsz * 2 * Hsz);
}
